// VisualContextFusion_10170482557443
// MI455X (gfx1250) — matrix operand order repaired and checked by reading, not compiled, not run on hardware
//
#include <hip/hip_runtime.h>

// ---------------------------------------------------------------------------
// Problem constants (reference setup: B=32, n=1024, d=768, k=10)
// ---------------------------------------------------------------------------
#define BB     32
#define NN     1024
#define DD     768
#define KK     10
#define MROWS  (BB * NN)          // 32768
#define TWOD   (2 * DD)           // 1536
#define PADINF 3000.0f

typedef __attribute__((ext_vector_type(16))) __bf16 v16bf;
typedef __attribute__((ext_vector_type(8)))  __bf16 v8bf;
typedef __attribute__((ext_vector_type(2)))  __bf16 v2bf;
typedef __attribute__((ext_vector_type(8)))  float  v8f;
typedef int v4i_vs __attribute__((vector_size(16)));   // matches builtin param

#define AS_GLOBAL __attribute__((address_space(1)))
#define AS_LDS    __attribute__((address_space(3)))

union Frag16 { v16bf v; v8bf h[2]; };

// Optional CDNA5 async global->LDS path (ASYNCcnt-tracked), feature-guarded.
#if defined(__HIP_DEVICE_COMPILE__) &&                                   \
    __has_builtin(__builtin_amdgcn_global_load_async_to_lds_b128) &&     \
    __has_builtin(__builtin_amdgcn_s_wait_asynccnt)
#define HAVE_ASYNC_LDS 1
#else
#define HAVE_ASYNC_LDS 0
#endif

__device__ __forceinline__ unsigned short f2bf(float f) {
  union { float f; unsigned u; } x; x.f = f;
  unsigned r = x.u + 0x7FFFu + ((x.u >> 16) & 1u);   // round-to-nearest-even
  return (unsigned short)(r >> 16);
}
__device__ __forceinline__ float bf2f(unsigned short h) {
  union { unsigned u; float f; } x; x.u = ((unsigned)h) << 16;
  return x.f;
}
// pack two f32 -> packed bf16x2 (one v_cvt_pk_bf16_f32 when available)
__device__ __forceinline__ unsigned pack_bf16(float a, float b) {
#if defined(__HIP_DEVICE_COMPILE__) && __has_builtin(__builtin_amdgcn_cvt_pk_bf16_f32)
  union { v2bf v; unsigned u; } x;
  x.v = __builtin_amdgcn_cvt_pk_bf16_f32(a, b);
  return x.u;
#else
  return (unsigned)f2bf(a) | ((unsigned)f2bf(b) << 16);
#endif
}

// ---------------------------------------------------------------------------
// K0: convert W (768 x 1536, f32) -> bf16, same layout (vectorized 4-wide)
// ---------------------------------------------------------------------------
__global__ void wconv_kernel(const float* __restrict__ W,
                             unsigned short* __restrict__ Wbf, int nQuads) {
  int i = blockIdx.x * 256 + threadIdx.x;
  if (i < nQuads) {
    float4 v = *(const float4*)&W[(size_t)i * 4];
    uint2 p;
    p.x = pack_bf16(v.x, v.y);
    p.y = pack_bf16(v.z, v.w);
    *(uint2*)&Wbf[(size_t)i * 4] = p;
  }
}

// ---------------------------------------------------------------------------
// K1: box centers with +INF shift for padded boxes
// ---------------------------------------------------------------------------
__global__ void centers_kernel(const float* __restrict__ sp,
                               const int* __restrict__ mask,
                               float* __restrict__ centers) {
  int m = blockIdx.x * 256 + threadIdx.x;          // exactly MROWS threads
  float x1 = sp[m * 6 + 0], y1 = sp[m * 6 + 1];
  float x2 = sp[m * 6 + 2], y2 = sp[m * 6 + 3];
  float add = (mask[m] != 0) ? 0.f : PADINF;
  centers[m * 2 + 0] = (x1 + x2) * 0.5f + add;
  centers[m * 2 + 1] = (y1 + y2) * 0.5f + add;
}

// ---------------------------------------------------------------------------
// K2: per-batch column sums S_b[d] = sum_i encI[b,i,d]
// ---------------------------------------------------------------------------
__global__ void rowsum_kernel(const float* __restrict__ encI,
                              float* __restrict__ S) {
  int b  = blockIdx.x / 3;
  int dd = (blockIdx.x % 3) * 256 + threadIdx.x;
  const float* p = encI + (size_t)b * NN * DD + dd;
  float s = 0.f;
  for (int i = 0; i < NN; ++i) s += p[(size_t)i * DD];
  S[b * DD + dd] = s;
}

// ---------------------------------------------------------------------------
// K3: s2[b,n] = S_b @ W2^T  (W2 = W[:, 768:1536])
// ---------------------------------------------------------------------------
__global__ void s2_kernel(const float* __restrict__ S,
                          const float* __restrict__ W,
                          float* __restrict__ s2) {
  int b  = blockIdx.x / 3;
  int nn = (blockIdx.x % 3) * 256 + threadIdx.x;
  const float* wrow = W + (size_t)nn * TWOD + DD;
  const float* sb   = S + b * DD;
  float a = 0.f;
  for (int k = 0; k < DD; ++k) a += sb[k] * wrow[k];
  s2[b * DD + nn] = a;
}

// ---------------------------------------------------------------------------
// K4: top-10 nearest centers per row. One wave32 per row.
// 64-bit key = (f32bits(d2) << 10) | j  -> min gives smallest dist, then
// smallest index (matches jax stable top_k tiebreak). Self -> sentinel.
// ---------------------------------------------------------------------------
__global__ void __launch_bounds__(256) topk_kernel(const float* __restrict__ centers,
                                                   int* __restrict__ nbrIdx) {
  const int lane = threadIdx.x & 31;
  const int wv   = threadIdx.x >> 5;
  const int m    = blockIdx.x * 8 + wv;            // 4096 blocks * 8 waves
  const int b    = m >> 10;
  const int i    = m & (NN - 1);
  const float* cb = centers + (size_t)b * NN * 2;
  const float cix = cb[2 * i], ciy = cb[2 * i + 1];

  unsigned long long kq[KK];
#pragma unroll
  for (int s = 0; s < KK; ++s) kq[s] = ~0ull;

  for (int c = 0; c < NN / 32; ++c) {
    int j = lane + c * 32;
    float dx = cb[2 * j] - cix, dy = cb[2 * j + 1] - ciy;
    float d2 = dx * dx + dy * dy;
    union { float f; unsigned u; } fb; fb.f = d2;
    unsigned long long key =
        (j == i) ? ~0ull
                 : ((((unsigned long long)fb.u) << 10) | (unsigned)j);
    if (key < kq[KK - 1]) {
      kq[KK - 1] = key;
#pragma unroll
      for (int s = KK - 1; s > 0; --s) {
        if (kq[s] < kq[s - 1]) {
          unsigned long long t = kq[s]; kq[s] = kq[s - 1]; kq[s - 1] = t;
        }
      }
    }
  }
  // 32-way sorted-list merge, 10 rounds
  for (int t = 0; t < KK; ++t) {
    unsigned long long mk = kq[0];
#pragma unroll
    for (int off = 16; off >= 1; off >>= 1) {
      unsigned long long o = __shfl_xor(mk, off, 32);
      if (o < mk) mk = o;
    }
    if (kq[0] == mk) {                         // unique owner (indices unique)
#pragma unroll
      for (int s = 0; s < KK - 1; ++s) kq[s] = kq[s + 1];
      kq[KK - 1] = ~0ull;
    }
    if (lane == 0) nbrIdx[m * 16 + t] = (int)(mk & 1023u);
  }
}

// ---------------------------------------------------------------------------
// K5: per-row softmax coefficients. One wave32 per row.
//   c_t = valid ? exp(sim_t - M) - e0 : 0,  Z = n*e0 + sum(c_t)
// store  nbrCoef = c_t / Z,  gscale = e0 / Z
// ---------------------------------------------------------------------------
__global__ void __launch_bounds__(256) coef_kernel(const float* __restrict__ encI,
                                                   const int* __restrict__ mask,
                                                   const int* __restrict__ nbrIdx,
                                                   float* __restrict__ nbrCoef,
                                                   float* __restrict__ gscale) {
  const int lane = threadIdx.x & 31;
  const int wv   = threadIdx.x >> 5;
  const int m    = blockIdx.x * 8 + wv;
  const int b    = m >> 10;
  const int i    = m & (NN - 1);

  const float* rowi = encI + (size_t)m * DD;
  float ai[DD / 32];
#pragma unroll
  for (int c = 0; c < DD / 32; ++c) ai[c] = rowi[lane + 32 * c];

  float sims[KK]; int vld[KK];
#pragma unroll
  for (int t = 0; t < KK; ++t) {
    int j = nbrIdx[m * 16 + t];
    vld[t] = (j != i) && (mask[b * NN + j] != 0);
    const float* rowj = encI + ((size_t)b * NN + j) * DD;
    float p = 0.f;
#pragma unroll
    for (int c = 0; c < DD / 32; ++c) p += ai[c] * rowj[lane + 32 * c];
#pragma unroll
    for (int off = 16; off >= 1; off >>= 1) p += __shfl_xor(p, off, 32);
    sims[t] = p;
  }
  float Mx = 0.f;
#pragma unroll
  for (int t = 0; t < KK; ++t) if (vld[t] && sims[t] > Mx) Mx = sims[t];
  float e0 = expf(-Mx);
  float ct[KK]; float cs = 0.f;
#pragma unroll
  for (int t = 0; t < KK; ++t) {
    ct[t] = vld[t] ? (expf(sims[t] - Mx) - e0) : 0.f;
    cs += ct[t];
  }
  float Z  = (float)NN * e0 + cs;
  float iz = 1.f / Z;
  if (lane == 0) {
#pragma unroll
    for (int t = 0; t < KK; ++t) nbrCoef[m * 16 + t] = ct[t] * iz;
    gscale[m] = e0 * iz;
  }
}

// ---------------------------------------------------------------------------
// K6: WMMA GEMM  (M=32768, N=768, K=768), A = encI (f32 -> bf16 in LDS),
// B = Wbf rows (woff selects W1 / W2 half). 128x128 block tile, 8 waves,
// each wave 32x64 via 2x4 v_wmma_f32_16x16x32_bf16. Double-buffered LDS,
// one barrier per k-step; B tile optionally via async global->LDS (ASYNCcnt).
//   mode 0: store P = encI @ W2^T as bf16
//   mode 1: out = acc + b + gscale*s2 + sum_t coef_t * P[j_t]   (f32)
// ---------------------------------------------------------------------------
#define LDT   40                  // skewed LDS row stride (halfs)
#define TSZ   (128 * LDT)         // one tile buffer, in halfs

__global__ void __launch_bounds__(256)
gemm_pass(const float* __restrict__ encI, const unsigned short* __restrict__ Wbf,
          int woff, int mode,
          const unsigned short* __restrict__ Pin, unsigned short* __restrict__ Pout,
          const float* __restrict__ bvec, const float* __restrict__ s2,
          const float* __restrict__ gscale, const int* __restrict__ nbrIdx,
          const float* __restrict__ nbrCoef, float* __restrict__ out) {
  __shared__ unsigned short lA[2 * TSZ];
  __shared__ unsigned short lB[2 * TSZ];

  const int tid  = threadIdx.x;
  const int m0   = blockIdx.x * 128;
  const int n0   = blockIdx.y * 128;
  const int lane = tid & 31;
  const int wv   = tid >> 5;
  const int wm   = wv & 3;            // 4 waves along M
  const int wn   = wv >> 2;           // 2 waves along N
  const int rsel = lane & 15;
  const int hi   = lane >> 4;         // half-wave select
  const int kA   = hi * 8;            // A frag: K {0..7,16..23} / {8..15,24..31}
  const int kB   = hi * 8;            // B frag: K {0..7,16..23} / {8..15,24..31}

  v8f acc[2][4];
#pragma unroll
  for (int a = 0; a < 2; ++a)
#pragma unroll
    for (int c = 0; c < 4; ++c)
#pragma unroll
      for (int r = 0; r < 8; ++r) acc[a][c][r] = 0.f;

  // ---- tile staging (vectorized; B optionally async global->LDS) ----
  auto stage = [&](int k0, int buf) {
    unsigned short* A  = lA + buf * TSZ;
    unsigned short* Bs = lB + buf * TSZ;
    // A: 128 rows x 32 k of f32 -> bf16; 8 float4-chunks per row
#pragma unroll
    for (int i = 0; i < 4; ++i) {
      int e = tid + 256 * i; int r = e >> 3; int c = e & 7;
      float4 v = *(const float4*)&encI[(size_t)(m0 + r) * DD + k0 + c * 4];
      uint2 p; p.x = pack_bf16(v.x, v.y); p.y = pack_bf16(v.z, v.w);
      *(uint2*)&A[r * LDT + c * 4] = p;
    }
    // B: 128 rows x 32 k bf16; 4 b128-chunks per row
#pragma unroll
    for (int i = 0; i < 2; ++i) {
      int e = tid + 256 * i; int r = e >> 2; int c = e & 3;
      const unsigned short* src = &Wbf[(size_t)(n0 + r) * TWOD + woff + k0 + c * 8];
      unsigned short* dst = &Bs[r * LDT + c * 8];
#if HAVE_ASYNC_LDS
      __builtin_amdgcn_global_load_async_to_lds_b128(
          (AS_GLOBAL v4i_vs*)src, (AS_LDS v4i_vs*)dst, 0, 0);
#else
      *(uint4*)dst = *(const uint4*)src;
#endif
    }
  };

  stage(0, 0);
#if HAVE_ASYNC_LDS
  __builtin_amdgcn_s_wait_asynccnt(0);
#endif
  __syncthreads();

  for (int k0 = 0; k0 < DD; k0 += 32) {
    const int buf = (k0 >> 5) & 1;
    if (k0 + 32 < DD) {
      __builtin_prefetch(&encI[(size_t)(m0 + (tid >> 1)) * DD + k0 + 64], 0, 1);
      stage(k0 + 32, buf ^ 1);
    }
    const unsigned short* A  = lA + buf * TSZ;
    const unsigned short* Bs = lB + buf * TSZ;

    Frag16 fa[2], fb[4];
#pragma unroll
    for (int tm = 0; tm < 2; ++tm) {
      int row = wm * 32 + tm * 16 + rsel;
      fa[tm].h[0] = *(const v8bf*)&A[row * LDT + kA];
      fa[tm].h[1] = *(const v8bf*)&A[row * LDT + 16 + kA];
    }
#pragma unroll
    for (int tn = 0; tn < 4; ++tn) {
      int row = wn * 64 + tn * 16 + rsel;
      fb[tn].h[0] = *(const v8bf*)&Bs[row * LDT + kB];
      fb[tn].h[1] = *(const v8bf*)&Bs[row * LDT + 16 + kB];
    }
#pragma unroll
    for (int tm = 0; tm < 2; ++tm)
#pragma unroll
      for (int tn = 0; tn < 4; ++tn)
        acc[tm][tn] = __builtin_amdgcn_wmma_f32_16x16x32_bf16(
            false, fa[tm].v, false, fb[tn].v, (short)0, acc[tm][tn],
            false, false);
#if HAVE_ASYNC_LDS
    __builtin_amdgcn_s_wait_asynccnt(0);
#endif
    __syncthreads();
  }

  const int batch = m0 >> 10;   // 128-row tiles never straddle a batch
  if (mode == 0) {
#pragma unroll
    for (int tm = 0; tm < 2; ++tm)
#pragma unroll
      for (int tn = 0; tn < 4; ++tn) {
        int bm   = m0 + wm * 32 + tm * 16 + hi * 8;
        int ncol = n0 + wn * 64 + tn * 16 + rsel;
#pragma unroll
        for (int r = 0; r < 8; ++r)
          Pout[(size_t)(bm + r) * DD + ncol] = f2bf(acc[tm][tn][r]);
      }
  } else {
    const float* s2b = s2 + batch * DD;
    const unsigned short* Pb = Pin + (size_t)batch * NN * DD;
#pragma unroll
    for (int tm = 0; tm < 2; ++tm)
#pragma unroll
      for (int tn = 0; tn < 4; ++tn) {
        int bm   = m0 + wm * 32 + tm * 16 + hi * 8;
        int ncol = n0 + wn * 64 + tn * 16 + rsel;
        float bn  = bvec[ncol];
        float s2v = s2b[ncol];
#pragma unroll
        for (int r = 0; r < 8; ++r) {
          int m = bm + r;
          float v = acc[tm][tn][r] + bn + gscale[m] * s2v;
#pragma unroll
          for (int t = 0; t < KK; ++t) {
            float cf = nbrCoef[m * 16 + t];
            int   j  = nbrIdx[m * 16 + t];
            v += cf * bf2f(Pb[(size_t)j * DD + ncol]);
          }
          out[(size_t)m * DD + ncol] = v;
        }
      }
  }
}

// ---------------------------------------------------------------------------
// Host launcher
// ---------------------------------------------------------------------------
extern "C" void kernel_launch(void* const* d_in, const int* in_sizes, int n_in,
                              void* d_out, int out_size, void* d_ws, size_t ws_size,
                              hipStream_t stream) {
  const float* encI = (const float*)d_in[0];
  const int*   mask = (const int*)d_in[1];
  const float* sp   = (const float*)d_in[2];
  const float* W    = (const float*)d_in[3];
  const float* bvec = (const float*)d_in[4];
  // d_in[5] = k (scalar, 10 in reference setup; hardcoded as KK)
  (void)in_sizes; (void)n_in; (void)out_size; (void)ws_size;

  char*  ws  = (char*)d_ws;
  size_t off = 0;
  auto alloc = [&](size_t bytes) -> void* {
    void* p = (void*)(ws + off);
    off += (bytes + 255) & ~(size_t)255;
    return p;
  };
  unsigned short* Wbf     = (unsigned short*)alloc((size_t)DD * TWOD * 2);   // 2.25 MB
  float*          centers = (float*)alloc((size_t)MROWS * 2 * 4);            // 256 KB
  float*          S       = (float*)alloc((size_t)BB * DD * 4);              // 96 KB
  float*          s2      = (float*)alloc((size_t)BB * DD * 4);              // 96 KB
  int*            nbrIdx  = (int*)alloc((size_t)MROWS * 16 * 4);             // 2 MB
  float*          nbrCoef = (float*)alloc((size_t)MROWS * 16 * 4);           // 2 MB
  float*          gscale  = (float*)alloc((size_t)MROWS * 4);                // 128 KB
  unsigned short* P       = (unsigned short*)alloc((size_t)MROWS * DD * 2);  // 48 MB
  float* out = (float*)d_out;

  wconv_kernel<<<(DD * TWOD / 4 + 255) / 256, 256, 0, stream>>>(W, Wbf, DD * TWOD / 4);
  centers_kernel<<<MROWS / 256, 256, 0, stream>>>(sp, mask, centers);
  rowsum_kernel<<<BB * 3, 256, 0, stream>>>(encI, S);
  s2_kernel<<<BB * 3, 256, 0, stream>>>(S, W, s2);
  topk_kernel<<<MROWS / 8, 256, 0, stream>>>(centers, nbrIdx);
  coef_kernel<<<MROWS / 8, 256, 0, stream>>>(encI, mask, nbrIdx, nbrCoef, gscale);

  dim3 grid(MROWS / 128, DD / 128);   // 256 x 6
  // pass 1: P = encI @ W2^T  (bf16)
  gemm_pass<<<grid, 256, 0, stream>>>(encI, Wbf, DD, 0,
                                      nullptr, P,
                                      nullptr, nullptr, nullptr, nullptr,
                                      nullptr, nullptr);
  // pass 2: out = encI @ W1^T + b + gscale*s2 + sum_t coef_t * P[j_t]
  gemm_pass<<<grid, 256, 0, stream>>>(encI, Wbf, 0, 1,
                                      P, nullptr,
                                      bvec, s2, gscale, nbrIdx,
                                      nbrCoef, out);
}
